// SpatialPyramidPoolingGELU_61022895341954
// MI455X (gfx1250) — matrix operand order repaired and checked by reading, not compiled, not run on hardware
//
#include <hip/hip_runtime.h>
#include <hip/hip_bf16.h>
#include <math.h>

// ---------------------------------------------------------------------------
// SpatialPyramidPoolingGELU for MI455X (gfx1250), wave32 + bf16 WMMA.
// LDS tiles laid out so every WMMA fragment is two aligned ds_load_b128.
// ---------------------------------------------------------------------------

typedef __attribute__((ext_vector_type(16))) __bf16 bf16x16;
typedef __attribute__((ext_vector_type(8)))  __bf16 bf16x8;
typedef __attribute__((ext_vector_type(8)))  float  f32x8;

#define EPSV 1e-6f

#define B_       8
#define CIN      512
#define H_       64
#define W_       128
#define HW       8192      // H_*W_
#define NPIX     65536     // B_*HW
#define BT       512
#define LVL_CH   128
#define FINAL_CH 1024
#define OUT_CH   128

#define SWK 40    // sW row stride (bf16): 80 B rows, 16B-aligned 8-elem chunks
#define SXK 40    // sX row stride (bf16): transposed [px][k] layout

__constant__ int dGH[4] = {6, 3, 2, 1};
__constant__ int dGW[4] = {12, 6, 4, 2};

__device__ inline float gelu_f(float x) {
    return 0.5f * x * (1.0f + erff(x * 0.70710678118654752440f));
}

// bilinear weights, align_corners=False (PyTorch semantics)
__device__ inline void binterp(int i, int n_in, int n_out, int& i0, int& i1, float& f) {
    float src = ((float)i + 0.5f) * (float)n_in / (float)n_out - 0.5f;
    src = fmaxf(src, 0.0f);
    i0 = (int)src;                 // src >= 0, truncation == floor
    i1 = i0 + 1;
    if (i1 > n_in - 1) i1 = n_in - 1;
    f = src - (float)i0;
}

// two aligned 16-byte LDS loads -> one v16bf fragment
__device__ inline bf16x16 ld_frag2(const __bf16* p0, const __bf16* p1) {
    bf16x8 a = *(const bf16x8*)p0;
    bf16x8 b = *(const bf16x8*)p1;
    return __builtin_shufflevector(a, b, 0, 1, 2, 3, 4, 5, 6, 7,
                                   8, 9, 10, 11, 12, 13, 14, 15);
}

// ---------------------------------------------------------------------------
// Shared WMMA micro-tile step: wave computes 64(o) x 32(px) = 4x2 frags.
// A frag (16x32 bf16): lane m=lane&15, k = (lane>>4)*8 + j (+16 for j>=8)
//   -> sW row m: chunks at k8 and 16+k8, each 16B aligned (SWK=40).
// B frag (32x16 bf16): lane n=lane&15, k = (lane>>4)*8 + j (+16 for j>=8)
//   -> sX row n (transposed layout): chunks at k8 and 16+k8, 16B aligned.
// ---------------------------------------------------------------------------
__device__ inline void wmma_step(const __bf16* __restrict__ sW,
                                 const __bf16* __restrict__ sX,
                                 int wo, int wp, int lane, f32x8 (&acc)[4][2]) {
    const int m15 = lane & 15;
    const int hi  = lane >> 4;
    bf16x16 af[4], bfr[2];
#pragma unroll
    for (int mt = 0; mt < 4; ++mt) {
        const __bf16* row = sW + (size_t)(wo * 64 + mt * 16 + m15) * SWK;
        af[mt] = ld_frag2(row + hi * 8, row + 16 + hi * 8);
    }
#pragma unroll
    for (int nt = 0; nt < 2; ++nt) {
        const __bf16* row = sX + (size_t)(wp * 32 + nt * 16 + m15) * SXK;
        bfr[nt] = ld_frag2(row + hi * 8, row + 16 + hi * 8);
    }
#pragma unroll
    for (int mt = 0; mt < 4; ++mt)
#pragma unroll
        for (int nt = 0; nt < 2; ++nt)
            acc[mt][nt] = __builtin_amdgcn_wmma_f32_16x16x32_bf16(
                false, af[mt], false, bfr[nt], (short)0, acc[mt][nt], false, false);
}

// ---------------------------------------------------------------------------
// K1: per-pixel LayerNorm stats over the 512 input channels.
// ---------------------------------------------------------------------------
__global__ void stats0_kernel(const float* __restrict__ x,
                              float* __restrict__ mean0, float* __restrict__ rstd0) {
    int p  = blockIdx.x * blockDim.x + threadIdx.x;
    int b  = p >> 13;
    int hw = p & (HW - 1);
    const float* xb = x + (size_t)b * CIN * HW + hw;
    float s = 0.f, ss = 0.f;
    for (int c = 0; c < CIN; ++c) {
        float v = xb[(size_t)c * HW];
        s += v; ss += v * v;
    }
    float m   = s * (1.0f / CIN);
    float var = ss * (1.0f / CIN) - m * m;
    mean0[p] = m;
    rstd0[p] = rsqrtf(var + EPSV);
}

// ---------------------------------------------------------------------------
// K2: y = conv0( GELU( LN(x) ) )  — bf16 WMMA GEMM, M=512 N=65536 K=512.
// Block tile 128(o) x 128(px); px block == one image row.
// ---------------------------------------------------------------------------
__global__ __launch_bounds__(256) void gemm1_kernel(
    const float* __restrict__ x, const float* __restrict__ w0,
    const float* __restrict__ lnw, const float* __restrict__ lnb,
    const float* __restrict__ mean0, const float* __restrict__ rstd0,
    float* __restrict__ y) {
    __shared__ __align__(16) __bf16 sW[128 * SWK];
    __shared__ __align__(16) __bf16 sX[128 * SXK];
    __shared__ float sMean[128], sRstd[128];

    const int pxb = blockIdx.x & 511;   // 512 pixel blocks
    const int ob  = blockIdx.x >> 9;    // 4 output blocks
    const int o0  = ob * 128;
    const int b   = pxb >> 6;
    const int hw0 = (pxb & 63) * 128;
    const int p0  = b * HW + hw0;

    const int lane = threadIdx.x & 31;
    const int wave = threadIdx.x >> 5;
    const int wo = wave >> 2, wp = wave & 3;

    if (threadIdx.x < 128) {
        sMean[threadIdx.x] = mean0[p0 + threadIdx.x];
        sRstd[threadIdx.x] = rstd0[p0 + threadIdx.x];
    }

    f32x8 acc[4][2];
#pragma unroll
    for (int mt = 0; mt < 4; ++mt)
#pragma unroll
        for (int nt = 0; nt < 2; ++nt)
#pragma unroll
            for (int r = 0; r < 8; ++r) acc[mt][nt][r] = 0.f;

    const float* xb = x + (size_t)b * CIN * HW + hw0;

    for (int kb = 0; kb < CIN; kb += 32) {
        __syncthreads();
        // stage activations: task = (col, 8-wide k chunk); coalesced b32 global
        // reads, one aligned b128 LDS store per task. LN + GELU fused.
        for (int t = threadIdx.x; t < 512; t += 256) {
            int col = t & 127;
            int kr8 = (t >> 7) << 3;            // 0,8,16,24
            float mu = sMean[col], rs = sRstd[col];
            bf16x8 pk;
#pragma unroll
            for (int j = 0; j < 8; ++j) {
                int ch = kb + kr8 + j;
                float v = xb[(size_t)ch * HW + col];
                v = (v - mu) * rs * lnw[ch] + lnb[ch];
                pk[j] = (__bf16)gelu_f(v);
            }
            *(bf16x8*)(sX + (size_t)col * SXK + kr8) = pk;
        }
        // stage weights: 32B contiguous global reads, one b128 LDS store
        for (int t = threadIdx.x; t < 512; t += 256) {
            int orow = t >> 2;
            int kc8  = (t & 3) << 3;
            const float* src = w0 + (size_t)(o0 + orow) * CIN + kb + kc8;
            float4 a = *(const float4*)src;
            float4 c = *(const float4*)(src + 4);
            bf16x8 pk;
            pk[0] = (__bf16)a.x; pk[1] = (__bf16)a.y;
            pk[2] = (__bf16)a.z; pk[3] = (__bf16)a.w;
            pk[4] = (__bf16)c.x; pk[5] = (__bf16)c.y;
            pk[6] = (__bf16)c.z; pk[7] = (__bf16)c.w;
            *(bf16x8*)(sW + (size_t)orow * SWK + kc8) = pk;
        }
        if (kb + 32 < CIN)   // global_prefetch_b8 of next K tile
            __builtin_prefetch(&xb[(size_t)(kb + 32 + (threadIdx.x >> 3)) * HW +
                                   ((threadIdx.x & 7) << 4)], 0, 0);
        __syncthreads();
        wmma_step(sW, sX, wo, wp, lane, acc);
    }

#pragma unroll
    for (int mt = 0; mt < 4; ++mt) {
        int o = o0 + wo * 64 + mt * 16 + ((lane >> 4) * 8);
#pragma unroll
        for (int nt = 0; nt < 2; ++nt) {
            int px = wp * 32 + nt * 16 + (lane & 15);
            float* dst = y + ((size_t)b * BT + o) * HW + hw0 + px;
#pragma unroll
            for (int r = 0; r < 8; ++r) dst[(size_t)r * HW] = acc[mt][nt][r];
        }
    }
}

// ---------------------------------------------------------------------------
// K3: adaptive average pool (y fits in 192MB L2, so per-bin global reads hit L2)
// ---------------------------------------------------------------------------
__global__ void pool_kernel(const float* __restrict__ y, float* __restrict__ xp,
                            int gh, int gw, int n) {
    int idx = blockIdx.x * blockDim.x + threadIdx.x;
    if (idx >= n) return;
    int ghw = gh * gw;
    int f = idx % gw;
    int g = (idx / gw) % gh;
    int c = (idx / ghw) % CIN;
    int b = idx / (ghw * CIN);
    int hs  = (g * H_) / gh,  he = ((g + 1) * H_ + gh - 1) / gh;
    int ws_ = (f * W_) / gw,  we = ((f + 1) * W_ + gw - 1) / gw;
    const float* plane = y + ((size_t)b * CIN + c) * HW;
    float s = 0.f;
    for (int h = hs; h < he; ++h)
        for (int w = ws_; w < we; ++w) s += plane[h * W_ + w];
    xp[idx] = s / (float)((he - hs) * (we - ws_));
}

// ---------------------------------------------------------------------------
// K4: per-level LN + GELU + 1x1 conv 512->128 at pooled resolution.
// One wave per pooled pixel (grids are exact: npix % 4 == 0).
// ---------------------------------------------------------------------------
__global__ __launch_bounds__(128) void lvlconv_kernel(
    const float* __restrict__ xp, const float* __restrict__ cw,
    const float* __restrict__ lnw, const float* __restrict__ lnb,
    float* __restrict__ lvl, int ghw) {
    __shared__ float svec[4 * BT];
    int wave = threadIdx.x >> 5, lane = threadIdx.x & 31;
    int pix = blockIdx.x * 4 + wave;
    int b = pix / ghw, gf = pix % ghw;
    const float* xin = xp + (size_t)b * CIN * ghw + gf;

    float rv[16];
    float s = 0.f, ss = 0.f;
#pragma unroll
    for (int j = 0; j < 16; ++j) {
        int c = lane + j * 32;
        float v = xin[(size_t)c * ghw];
        rv[j] = v; s += v; ss += v * v;
    }
#pragma unroll
    for (int m = 16; m >= 1; m >>= 1) {     // wave32 butterfly reduction
        s  += __shfl_xor(s,  m, 32);
        ss += __shfl_xor(ss, m, 32);
    }
    float mean = s * (1.0f / CIN);
    float rstd = rsqrtf(ss * (1.0f / CIN) - mean * mean + EPSV);

    float* vec = svec + wave * BT;
#pragma unroll
    for (int j = 0; j < 16; ++j) {
        int c = lane + j * 32;
        float v = (rv[j] - mean) * rstd * lnw[c] + lnb[c];
        vec[c] = gelu_f(v);
    }
    __syncthreads();
    for (int oo = 0; oo < 4; ++oo) {
        int o = lane + oo * 32;
        const float* wrow = cw + (size_t)o * CIN;
        float a = 0.f;
        for (int c = 0; c < CIN; ++c) a += vec[c] * wrow[c];
        lvl[((size_t)b * LVL_CH + o) * ghw + gf] = a;
    }
}

// ---------------------------------------------------------------------------
// K5: LN stats over the 1024 virtual concat channels (interp fused, levels
// are tiny and cache-resident; upsampled planes never materialized).
// ---------------------------------------------------------------------------
__global__ void statsF_kernel(const float* __restrict__ y,
                              const float* __restrict__ lvl0, const float* __restrict__ lvl1,
                              const float* __restrict__ lvl2, const float* __restrict__ lvl3,
                              float* __restrict__ meanF, float* __restrict__ rstdF) {
    int p  = blockIdx.x * blockDim.x + threadIdx.x;
    int b  = p >> 13;
    int hw = p & (HW - 1);
    int h  = hw >> 7, w = hw & 127;
    const float* yb = y + (size_t)b * BT * HW + hw;
    float s = 0.f, ss = 0.f;
    for (int c = 0; c < BT; ++c) {
        float v = yb[(size_t)c * HW];
        s += v; ss += v * v;
    }
    const float* lv[4] = {lvl0, lvl1, lvl2, lvl3};
#pragma unroll
    for (int l = 0; l < 4; ++l) {
        int gh = dGH[l], gw = dGW[l], ghw = gh * gw;
        int r0, r1, c0, c1; float fh, fw;
        binterp(h, gh, H_, r0, r1, fh);
        binterp(w, gw, W_, c0, c1, fw);
        float w00 = (1.f - fh) * (1.f - fw), w01 = (1.f - fh) * fw;
        float w10 = fh * (1.f - fw),         w11 = fh * fw;
        int i00 = r0 * gw + c0, i01 = r0 * gw + c1;
        int i10 = r1 * gw + c0, i11 = r1 * gw + c1;
        const float* base = lv[l] + (size_t)b * LVL_CH * ghw;
        for (int o = 0; o < LVL_CH; ++o) {
            const float* L = base + (size_t)o * ghw;
            float v = w00 * L[i00] + w01 * L[i01] + w10 * L[i10] + w11 * L[i11];
            s += v; ss += v * v;
        }
    }
    float m = s * (1.0f / FINAL_CH);
    meanF[p] = m;
    rstdF[p] = rsqrtf(ss * (1.0f / FINAL_CH) - m * m + EPSV);
}

// ---------------------------------------------------------------------------
// K6: out = convf( GELU( LN(cat) ) ) — WMMA GEMM, M=128 N=65536 K=1024.
// Channels >=512 of the B tile (sX) are interpolated on the fly from lvl arrays.
// ---------------------------------------------------------------------------
__global__ __launch_bounds__(256) void gemm2_kernel(
    const float* __restrict__ y, const float* __restrict__ wf,
    const float* __restrict__ lnw, const float* __restrict__ lnb,
    const float* __restrict__ meanF, const float* __restrict__ rstdF,
    const float* __restrict__ lvl0, const float* __restrict__ lvl1,
    const float* __restrict__ lvl2, const float* __restrict__ lvl3,
    float* __restrict__ out) {
    __shared__ __align__(16) __bf16 sW[128 * SWK];
    __shared__ __align__(16) __bf16 sX[128 * SXK];
    __shared__ float sMean[128], sRstd[128];

    const int pxb = blockIdx.x;        // 0..511, one image row per block
    const int b   = pxb >> 6;
    const int h   = pxb & 63;
    const int hw0 = h * 128;
    const int p0  = b * HW + hw0;
    const int lane = threadIdx.x & 31, wave = threadIdx.x >> 5;
    const int wo = wave >> 2, wp = wave & 3;

    if (threadIdx.x < 128) {
        sMean[threadIdx.x] = meanF[p0 + threadIdx.x];
        sRstd[threadIdx.x] = rstdF[p0 + threadIdx.x];
    }

    const float* lv[4] = {lvl0, lvl1, lvl2, lvl3};
    const float* yb = y + (size_t)b * BT * HW + hw0;

    f32x8 acc[4][2];
#pragma unroll
    for (int mt = 0; mt < 4; ++mt)
#pragma unroll
        for (int nt = 0; nt < 2; ++nt)
#pragma unroll
            for (int r = 0; r < 8; ++r) acc[mt][nt][r] = 0.f;

    for (int kb = 0; kb < FINAL_CH; kb += 32) {
        __syncthreads();
        if (kb < BT) {
            for (int t = threadIdx.x; t < 512; t += 256) {
                int col = t & 127;
                int kr8 = (t >> 7) << 3;
                float mu = sMean[col], rs = sRstd[col];
                bf16x8 pk;
#pragma unroll
                for (int j = 0; j < 8; ++j) {
                    int ch = kb + kr8 + j;
                    float v = yb[(size_t)ch * HW + col];
                    v = (v - mu) * rs * lnw[ch] + lnb[ch];
                    pk[j] = (__bf16)gelu_f(v);
                }
                *(bf16x8*)(sX + (size_t)col * SXK + kr8) = pk;
            }
        } else {
            int l  = (kb - BT) >> 7;             // 32-chunk never crosses a level
            int gh = dGH[l], gw = dGW[l], ghw = gh * gw;
            int r0, r1; float fh;
            binterp(h, gh, H_, r0, r1, fh);      // row weights uniform per block
            int ob_ = (kb - BT) & 127;
            const float* base = lv[l] + (size_t)b * LVL_CH * ghw;
            for (int t = threadIdx.x; t < 512; t += 256) {
                int col = t & 127;
                int kr8 = (t >> 7) << 3;
                int c0, c1; float fw;
                binterp(col, gw, W_, c0, c1, fw);
                float mu = sMean[col], rs = sRstd[col];
                bf16x8 pk;
#pragma unroll
                for (int j = 0; j < 8; ++j) {
                    const float* L = base + (size_t)(ob_ + kr8 + j) * ghw;
                    float v =
                        (1.f - fh) * ((1.f - fw) * L[r0 * gw + c0] + fw * L[r0 * gw + c1]) +
                        fh         * ((1.f - fw) * L[r1 * gw + c0] + fw * L[r1 * gw + c1]);
                    int ch = kb + kr8 + j;
                    v = (v - mu) * rs * lnw[ch] + lnb[ch];
                    pk[j] = (__bf16)gelu_f(v);
                }
                *(bf16x8*)(sX + (size_t)col * SXK + kr8) = pk;
            }
        }
        for (int t = threadIdx.x; t < 512; t += 256) {
            int orow = t >> 2;
            int kc8  = (t & 3) << 3;
            const float* src = wf + (size_t)orow * FINAL_CH + kb + kc8;
            float4 a = *(const float4*)src;
            float4 c = *(const float4*)(src + 4);
            bf16x8 pk;
            pk[0] = (__bf16)a.x; pk[1] = (__bf16)a.y;
            pk[2] = (__bf16)a.z; pk[3] = (__bf16)a.w;
            pk[4] = (__bf16)c.x; pk[5] = (__bf16)c.y;
            pk[6] = (__bf16)c.z; pk[7] = (__bf16)c.w;
            *(bf16x8*)(sW + (size_t)orow * SWK + kc8) = pk;
        }
        __syncthreads();
        wmma_step(sW, sX, wo, wp, lane, acc);
    }

#pragma unroll
    for (int mt = 0; mt < 4; ++mt) {
        int o = wo * 64 + mt * 16 + ((lane >> 4) * 8);
#pragma unroll
        for (int nt = 0; nt < 2; ++nt) {
            int px = wp * 32 + nt * 16 + (lane & 15);
            float* dst = out + ((size_t)b * OUT_CH + o) * HW + hw0 + px;
#pragma unroll
            for (int r = 0; r < 8; ++r) dst[(size_t)r * HW] = acc[mt][nt][r];
        }
    }
}

// ---------------------------------------------------------------------------
extern "C" void kernel_launch(void* const* d_in, const int* in_sizes, int n_in,
                              void* d_out, int out_size, void* d_ws, size_t ws_size,
                              hipStream_t stream) {
    (void)in_sizes; (void)n_in; (void)out_size; (void)ws_size;

    const float* x    = (const float*)d_in[0];
    const float* lnw0 = (const float*)d_in[1];
    const float* lnb0 = (const float*)d_in[2];
    const float* w0   = (const float*)d_in[3];
    const float* lnwf = (const float*)d_in[4];
    const float* lnbf = (const float*)d_in[5];
    const float* wfin = (const float*)d_in[6];
    const float *lnw_l[4], *lnb_l[4], *cw_l[4];
    for (int i = 0; i < 4; ++i) {
        lnw_l[i] = (const float*)d_in[7 + 3 * i];
        lnb_l[i] = (const float*)d_in[8 + 3 * i];
        cw_l[i]  = (const float*)d_in[9 + 3 * i];
    }

    const int GH[4] = {6, 3, 2, 1}, GW[4] = {12, 6, 4, 2};

    // workspace carve-out (~137 MB of floats)
    float* ws = (float*)d_ws;
    size_t off = 0;
    float* mean0 = ws + off; off += NPIX;
    float* rstd0 = ws + off; off += NPIX;
    float* y     = ws + off; off += (size_t)B_ * BT * HW;
    float* xp[4]; float* lvl[4];
    for (int l = 0; l < 4; ++l) { xp[l]  = ws + off; off += (size_t)B_ * CIN    * GH[l] * GW[l]; }
    for (int l = 0; l < 4; ++l) { lvl[l] = ws + off; off += (size_t)B_ * LVL_CH * GH[l] * GW[l]; }
    float* meanF = ws + off; off += NPIX;
    float* rstdF = ws + off; off += NPIX;

    stats0_kernel<<<NPIX / 256, 256, 0, stream>>>(x, mean0, rstd0);
    gemm1_kernel<<<2048, 256, 0, stream>>>(x, w0, lnw0, lnb0, mean0, rstd0, y);
    for (int l = 0; l < 4; ++l) {
        int n = B_ * CIN * GH[l] * GW[l];
        pool_kernel<<<(n + 255) / 256, 256, 0, stream>>>(y, xp[l], GH[l], GW[l], n);
    }
    for (int l = 0; l < 4; ++l) {
        int ghw = GH[l] * GW[l];
        lvlconv_kernel<<<(B_ * ghw) / 4, 128, 0, stream>>>(
            xp[l], cw_l[l], lnw_l[l], lnb_l[l], lvl[l], ghw);
    }
    statsF_kernel<<<NPIX / 256, 256, 0, stream>>>(y, lvl[0], lvl[1], lvl[2], lvl[3],
                                                  meanF, rstdF);
    gemm2_kernel<<<512, 256, 0, stream>>>(y, wfin, lnwf, lnbf, meanF, rstdF,
                                          lvl[0], lvl[1], lvl[2], lvl[3], (float*)d_out);
}
